// EncoderCUGOSum_55559696941462
// MI455X (gfx1250) — matrix operand order repaired and checked by reading, not compiled, not run on hardware
//
#include <hip/hip_runtime.h>
#include <hip/hip_fp16.h>

typedef __attribute__((ext_vector_type(16))) _Float16 v16h;
typedef __attribute__((ext_vector_type(8)))  _Float16 v8h;
typedef __attribute__((ext_vector_type(8)))  float    v8f;
typedef __attribute__((ext_vector_type(4)))  float    v4f;

#define DK 512
#define N_GRID 131072
#define N_MESH 10242
#define N_EDGE 262144

constexpr int BM = 128;   // rows per block
constexpr int BN = 128;   // cols per block
constexpr int BK = 32;    // k-slab
constexpr int AST = 40;   // As row stride (f16): 80B rows, 16B aligned
constexpr int BST = 136;  // Bs row stride (f16): 272B rows, 16B aligned

__device__ __forceinline__ float silu_f(float x) { return x / (1.0f + __expf(-x)); }

// ---------------- utility kernels ----------------
__global__ void pack_f16_kernel(const float* __restrict__ s, _Float16* __restrict__ d, int n) {
  int i = blockIdx.x * blockDim.x + threadIdx.x;
  int st = gridDim.x * blockDim.x;
  for (; i < n; i += st) d[i] = (_Float16)s[i];
}

__global__ void zero_f32_kernel(float* __restrict__ p, int n) {
  int i = blockIdx.x * blockDim.x + threadIdx.x;
  int st = gridDim.x * blockDim.x;
  for (; i < n; i += st) p[i] = 0.0f;
}

// ---------------- main WMMA GEMM ----------------
// C[M,512] = A[M,K] @ B[K,512]  (+ fused epilogue)
// 8 waves: 4 in M (32 rows each = 2 subtiles) x 2 in N (64 cols each = 4 subtiles).
// Double-buffered LDS: one barrier per K-slab, global loads overlap WMMA.
// AMODE: 0 = A is f32, 1 = A is f16, 2 = dual f32 sources split at k=512 (K=1024)
// EPI:   0 = f32 store (acc + bias)
//        1 = edge: acc + bias + srcp[src_idx[m]] + dstp[dst_idx[m]], silu, f16 store
//        2 = silu(acc + bias), f16 store
//        3 = acc + bias, f16 store
template <int AMODE, int EPI>
__global__ __launch_bounds__(256, 1)
void gemm_wmma_kernel(const void* __restrict__ Aany, const float* __restrict__ A2,
                      const _Float16* __restrict__ B, const float* __restrict__ bias,
                      void* __restrict__ Outany, int M, int K,
                      const float* __restrict__ srcp, const float* __restrict__ dstp,
                      const int* __restrict__ src_idx, const int* __restrict__ dst_idx) {
  __shared__ _Float16 As[2][BM * AST];
  __shared__ _Float16 Bs[2][BK * BST];

  const int tid   = threadIdx.x;
  const int wave  = tid >> 5;
  const int lane  = tid & 31;
  const int mwave = wave & 3;   // 0..3 -> 32-row strip
  const int nwave = wave >> 2;  // 0..1 -> 64-col strip
  const int tileM = blockIdx.x * BM;
  const int tileN = blockIdx.y * BN;

  v8f acc[2][4];
#pragma unroll
  for (int mt = 0; mt < 2; ++mt)
#pragma unroll
    for (int nt = 0; nt < 4; ++nt)
#pragma unroll
      for (int j = 0; j < 8; ++j) acc[mt][nt][j] = 0.0f;

  // global->LDS staging assignments
  const int arow = tid >> 1;            // 0..127
  const int acol = (tid & 1) * 16;      // 0 or 16
  const int grow = tileM + arow;
  const int brow = tid >> 3;            // 0..31
  const int bcol = (tid & 7) * 16;      // 0..112

  auto stage = [&](int k0, int buf) {
    // ---- A tile (BM x BK) ----
    _Float16* adst = As[buf] + arow * AST + acol;
    if (grow < M) {
      if (AMODE == 1) {
        const _Float16* a = (const _Float16*)Aany + (size_t)grow * K + k0 + acol;
        *(v8h*)adst       = *(const v8h*)a;
        *(v8h*)(adst + 8) = *(const v8h*)(a + 8);
      } else {
        const float* a;
        if (AMODE == 2) {
          int kk = k0 + acol;
          a = (kk < 512) ? ((const float*)Aany + (size_t)grow * 512 + kk)
                         : (A2 + (size_t)grow * 512 + (kk - 512));
        } else {
          a = (const float*)Aany + (size_t)grow * K + k0 + acol;
        }
        v4f f0 = *(const v4f*)a;
        v4f f1 = *(const v4f*)(a + 4);
        v4f f2 = *(const v4f*)(a + 8);
        v4f f3 = *(const v4f*)(a + 12);
        v8h h0, h1;
#pragma unroll
        for (int j = 0; j < 4; ++j) {
          h0[j] = (_Float16)f0[j]; h0[4 + j] = (_Float16)f1[j];
          h1[j] = (_Float16)f2[j]; h1[4 + j] = (_Float16)f3[j];
        }
        *(v8h*)adst = h0; *(v8h*)(adst + 8) = h1;
      }
    } else {
      v8h z;
#pragma unroll
      for (int j = 0; j < 8; ++j) z[j] = (_Float16)0.0f;
      *(v8h*)adst = z; *(v8h*)(adst + 8) = z;
    }
    // ---- B tile (BK x BN), weights already f16 ----
    const _Float16* b = B + (size_t)(k0 + brow) * DK + tileN + bcol;
    _Float16* bdst = Bs[buf] + brow * BST + bcol;
    *(v8h*)bdst       = *(const v8h*)b;
    *(v8h*)(bdst + 8) = *(const v8h*)(b + 8);
  };

  stage(0, 0);
  __syncthreads();
  int p = 0;
  for (int k0 = 0; k0 < K; k0 += BK) {
    if (k0 + BK < K) stage(k0 + BK, p ^ 1);  // overlap next slab with compute

    // A fragments (ISA layout: lane holds row lane&15, K = kb..kb+7, kb+16..kb+23)
    const int kb = (lane >> 4) * 8;
    v16h afr[2];
#pragma unroll
    for (int mt = 0; mt < 2; ++mt) {
      const _Float16* ap = As[p] + (mwave * 32 + mt * 16 + (lane & 15)) * AST + kb;
      v8h alo = *(const v8h*)ap;
      v8h ahi = *(const v8h*)(ap + 16);
#pragma unroll
      for (int j = 0; j < 8; ++j) { afr[mt][j] = alo[j]; afr[mt][8 + j] = ahi[j]; }
    }
#pragma unroll
    for (int nt = 0; nt < 4; ++nt) {
      // B fragment: lane = N-col (lane&15), K = kb..kb+7, kb+16..kb+23; reused across both M-subtiles
      const _Float16* bp = Bs[p] + kb * BST + nwave * 64 + nt * 16 + (lane & 15);
      v16h bfr;
#pragma unroll
      for (int j = 0; j < 8; ++j) { bfr[j] = bp[j * BST]; bfr[8 + j] = bp[(16 + j) * BST]; }
      acc[0][nt] = __builtin_amdgcn_wmma_f32_16x16x32_f16(
          false, afr[0], false, bfr, (short)0, acc[0][nt], false, false);
      acc[1][nt] = __builtin_amdgcn_wmma_f32_16x16x32_f16(
          false, afr[1], false, bfr, (short)0, acc[1][nt], false, false);
    }
    __syncthreads();
    p ^= 1;
  }

  // ---- epilogue ----
  // D layout: lane<16 -> N=lane, M=r ; lane>=16 -> N=lane-16, M=r+8
  const int nloc = lane & 15;
  int   cols[4];
  float bv[4];
#pragma unroll
  for (int nt = 0; nt < 4; ++nt) {
    cols[nt] = tileN + nwave * 64 + nt * 16 + nloc;
    bv[nt]   = bias ? bias[cols[nt]] : 0.0f;
  }

#pragma unroll
  for (int mt = 0; mt < 2; ++mt) {
    const int mo = tileM + mwave * 32 + mt * 16 + (lane >> 4) * 8;
#pragma unroll
    for (int r = 0; r < 8; ++r) {
      int row = mo + r;
      if (row >= M) continue;
      if (EPI == 1) {
        int si = src_idx[row];
        int di = dst_idx[row];
        const float* sp = srcp + (size_t)si * DK;
        const float* dp = dstp + (size_t)di * DK;
        _Float16* out = (_Float16*)Outany + (size_t)row * DK;
#pragma unroll
        for (int nt = 0; nt < 4; ++nt) {
          float v = acc[mt][nt][r] + bv[nt] + sp[cols[nt]] + dp[cols[nt]];
          out[cols[nt]] = (_Float16)silu_f(v);
        }
      } else if (EPI == 2) {
        _Float16* out = (_Float16*)Outany + (size_t)row * DK;
#pragma unroll
        for (int nt = 0; nt < 4; ++nt)
          out[cols[nt]] = (_Float16)silu_f(acc[mt][nt][r] + bv[nt]);
      } else if (EPI == 3) {
        _Float16* out = (_Float16*)Outany + (size_t)row * DK;
#pragma unroll
        for (int nt = 0; nt < 4; ++nt)
          out[cols[nt]] = (_Float16)(acc[mt][nt][r] + bv[nt]);
      } else {
        float* out = (float*)Outany + (size_t)row * DK;
#pragma unroll
        for (int nt = 0; nt < 4; ++nt)
          out[cols[nt]] = acc[mt][nt][r] + bv[nt];
      }
    }
  }
}

// ---------------- LayerNorm + atomic scatter-sum (edge rows -> agg[dst]) ----------------
__global__ __launch_bounds__(256)
void ln_scatter_kernel(const _Float16* __restrict__ t, const float* __restrict__ g,
                       const float* __restrict__ b, const int* __restrict__ dst_idx,
                       float* __restrict__ agg, int M) {
  int row = blockIdx.x;
  if (row >= M) return;
  const _Float16* tp = t + (size_t)row * DK;
  int tid = threadIdx.x;
  float x0 = (float)tp[tid];
  float x1 = (float)tp[tid + 256];
  float s = x0 + x1, q = x0 * x0 + x1 * x1;
#pragma unroll
  for (int o = 16; o > 0; o >>= 1) { s += __shfl_xor(s, o, 32); q += __shfl_xor(q, o, 32); }
  __shared__ float ssum[8], ssq[8];
  int wave = tid >> 5, lane = tid & 31;
  if (lane == 0) { ssum[wave] = s; ssq[wave] = q; }
  __syncthreads();
  if (tid == 0) {
    float S = 0.0f, Q = 0.0f;
    for (int i = 0; i < 8; ++i) { S += ssum[i]; Q += ssq[i]; }
    ssum[0] = S; ssq[0] = Q;
  }
  __syncthreads();
  float mu  = ssum[0] * (1.0f / 512.0f);
  float var = ssq[0] * (1.0f / 512.0f) - mu * mu;
  float rs  = rsqrtf(var + 1e-5f);
  int d = dst_idx[row];
  float* ap = agg + (size_t)d * DK;
  float y0 = (x0 - mu) * rs * g[tid] + b[tid];
  float y1 = (x1 - mu) * rs * g[tid + 256] + b[tid + 256];
  unsafeAtomicAdd(ap + tid, y0);
  unsafeAtomicAdd(ap + tid + 256, y1);
}

// ---------------- LayerNorm + residual: out = base + LN(t)*g + b ----------------
__global__ __launch_bounds__(256)
void ln_residual_kernel(const float* __restrict__ t, const float* __restrict__ g,
                        const float* __restrict__ b, const float* __restrict__ base,
                        float* __restrict__ out, int M) {
  int row = blockIdx.x;
  if (row >= M) return;
  const float* tp = t + (size_t)row * DK;
  int tid = threadIdx.x;
  float x0 = tp[tid];
  float x1 = tp[tid + 256];
  float s = x0 + x1, q = x0 * x0 + x1 * x1;
#pragma unroll
  for (int o = 16; o > 0; o >>= 1) { s += __shfl_xor(s, o, 32); q += __shfl_xor(q, o, 32); }
  __shared__ float ssum[8], ssq[8];
  int wave = tid >> 5, lane = tid & 31;
  if (lane == 0) { ssum[wave] = s; ssq[wave] = q; }
  __syncthreads();
  if (tid == 0) {
    float S = 0.0f, Q = 0.0f;
    for (int i = 0; i < 8; ++i) { S += ssum[i]; Q += ssq[i]; }
    ssum[0] = S; ssq[0] = Q;
  }
  __syncthreads();
  float mu  = ssum[0] * (1.0f / 512.0f);
  float var = ssq[0] * (1.0f / 512.0f) - mu * mu;
  float rs  = rsqrtf(var + 1e-5f);
  const float* bp = base + (size_t)row * DK;
  float* op = out + (size_t)row * DK;
  op[tid]       = bp[tid]       + (x0 - mu) * rs * g[tid]       + b[tid];
  op[tid + 256] = bp[tid + 256] + (x1 - mu) * rs * g[tid + 256] + b[tid + 256];
}

extern "C" void kernel_launch(void* const* d_in, const int* in_sizes, int n_in,
                              void* d_out, int out_size, void* d_ws, size_t ws_size,
                              hipStream_t stream) {
  (void)in_sizes; (void)n_in; (void)out_size; (void)ws_size;
  const float* g2m   = (const float*)d_in[0];
  const float* gridf = (const float*)d_in[1];
  const float* meshf = (const float*)d_in[2];
  const float* eW_e  = (const float*)d_in[3];
  const float* eW_s  = (const float*)d_in[4];
  const float* eW_d  = (const float*)d_in[5];
  const float* eb1   = (const float*)d_in[6];
  const float* eW2   = (const float*)d_in[7];
  const float* eb2   = (const float*)d_in[8];
  const float* eln_g = (const float*)d_in[9];
  const float* eln_b = (const float*)d_in[10];
  const float* sW1   = (const float*)d_in[11];
  const float* sb1   = (const float*)d_in[12];
  const float* sW2   = (const float*)d_in[13];
  const float* sb2   = (const float*)d_in[14];
  const float* sln_g = (const float*)d_in[15];
  const float* sln_b = (const float*)d_in[16];
  const float* dW1   = (const float*)d_in[17];
  const float* db1   = (const float*)d_in[18];
  const float* dW2   = (const float*)d_in[19];
  const float* db2   = (const float*)d_in[20];
  const float* dln_g = (const float*)d_in[21];
  const float* dln_b = (const float*)d_in[22];
  const int* src_idx = (const int*)d_in[23];
  const int* dst_idx = (const int*)d_in[24];

  // ---- workspace layout ----
  char* ws = (char*)d_ws;
  _Float16* Wh   = (_Float16*)ws;          // packed f16 weights
  _Float16* heWs = Wh + 0;
  _Float16* heWd = Wh + 262144;
  _Float16* heWe = Wh + 524288;
  _Float16* heW2 = Wh + 786432;
  _Float16* hsW1 = Wh + 1048576;
  _Float16* hsW2 = Wh + 1310720;
  _Float16* hdW1 = Wh + 1572864;           // [1024,512]
  _Float16* hdW2 = Wh + 2097152;
  size_t off = 2359296ull * 2;             // 4,718,592 B
  float*    srcp  = (float*)(ws + off);    off += (size_t)N_GRID * DK * 4;   // 256 MiB
  float*    dstp  = (float*)(ws + off);    off += (size_t)N_MESH * DK * 4;
  _Float16* hsilu = (_Float16*)(ws + off); off += (size_t)N_EDGE * DK * 2;   // 256 MiB
  float*    agg   = (float*)(ws + off);    off += (size_t)N_MESH * DK * 4;
  _Float16* hmesh = (_Float16*)(ws + off); off += (size_t)N_MESH * DK * 2;
  // aliases (safe by stream ordering)
  _Float16* tE = (_Float16*)srcp;          // edge pre-LN (f16) reuses src_proj region
  float*    t2 = dstp;                     // mesh pre-LN reuses dst_proj region
  _Float16* h1 = hsilu;                    // grid hidden reuses hsilu region
  float*    t3 = srcp;                     // grid pre-LN reuses src_proj region

  float* grid_out = (float*)d_out;
  float* mesh_out = grid_out + (size_t)N_GRID * DK;

  dim3 blk(256);
  // pack weights to f16
  pack_f16_kernel<<<512, 256, 0, stream>>>(eW_s, heWs, 262144);
  pack_f16_kernel<<<512, 256, 0, stream>>>(eW_d, heWd, 262144);
  pack_f16_kernel<<<512, 256, 0, stream>>>(eW_e, heWe, 262144);
  pack_f16_kernel<<<512, 256, 0, stream>>>(eW2,  heW2, 262144);
  pack_f16_kernel<<<512, 256, 0, stream>>>(sW1,  hsW1, 262144);
  pack_f16_kernel<<<512, 256, 0, stream>>>(sW2,  hsW2, 262144);
  pack_f16_kernel<<<512, 256, 0, stream>>>(dW1,  hdW1, 524288);
  pack_f16_kernel<<<512, 256, 0, stream>>>(dW2,  hdW2, 262144);
  zero_f32_kernel<<<512, 256, 0, stream>>>(agg, N_MESH * DK);

  auto gb = [](int M) { return dim3((unsigned)((M + BM - 1) / BM), 512 / BN); };

  // 1) src_proj = grid @ eW_s           (f32 out)
  gemm_wmma_kernel<0, 0><<<gb(N_GRID), blk, 0, stream>>>(
      gridf, nullptr, heWs, nullptr, srcp, N_GRID, DK, nullptr, nullptr, nullptr, nullptr);
  // 2) dst_proj = mesh @ eW_d           (f32 out)
  gemm_wmma_kernel<0, 0><<<gb(N_MESH), blk, 0, stream>>>(
      meshf, nullptr, heWd, nullptr, dstp, N_MESH, DK, nullptr, nullptr, nullptr, nullptr);
  // 3) hsilu = silu(g2m@eW_e + gather(srcp) + gather(dstp) + eb1)   (f16 out)
  gemm_wmma_kernel<0, 1><<<gb(N_EDGE), blk, 0, stream>>>(
      g2m, nullptr, heWe, eb1, hsilu, N_EDGE, DK, srcp, dstp, src_idx, dst_idx);
  // 4) tE = hsilu @ eW2 + eb2           (f16 out, into srcp region)
  gemm_wmma_kernel<1, 3><<<gb(N_EDGE), blk, 0, stream>>>(
      hsilu, nullptr, heW2, eb2, tE, N_EDGE, DK, nullptr, nullptr, nullptr, nullptr);
  // 5) agg[dst] += LN(tE)*g+b  (segment_sum fused with LayerNorm)
  ln_scatter_kernel<<<N_EDGE, 256, 0, stream>>>(tE, eln_g, eln_b, dst_idx, agg, N_EDGE);
  // 6) hmesh = silu([mesh|agg] @ dW1 + db1)   (dual-A, K=1024, f16 out)
  gemm_wmma_kernel<2, 2><<<gb(N_MESH), blk, 0, stream>>>(
      meshf, agg, hdW1, db1, hmesh, N_MESH, 1024, nullptr, nullptr, nullptr, nullptr);
  // 7) t2 = hmesh @ dW2 + db2           (f32 out)
  gemm_wmma_kernel<1, 0><<<gb(N_MESH), blk, 0, stream>>>(
      hmesh, nullptr, hdW2, db2, t2, N_MESH, DK, nullptr, nullptr, nullptr, nullptr);
  // 8) mesh_out = mesh + LN(t2)*g+b
  ln_residual_kernel<<<N_MESH, 256, 0, stream>>>(t2, dln_g, dln_b, meshf, mesh_out, N_MESH);
  // 9) h1 = silu(grid @ sW1 + sb1)      (f16 out, into hsilu region)
  gemm_wmma_kernel<0, 2><<<gb(N_GRID), blk, 0, stream>>>(
      gridf, nullptr, hsW1, sb1, h1, N_GRID, DK, nullptr, nullptr, nullptr, nullptr);
  // 10) t3 = h1 @ sW2 + sb2             (f32 out, into srcp region)
  gemm_wmma_kernel<1, 0><<<gb(N_GRID), blk, 0, stream>>>(
      h1, nullptr, hsW2, sb2, t3, N_GRID, DK, nullptr, nullptr, nullptr, nullptr);
  // 11) grid_out = grid + LN(t3)*g+b
  ln_residual_kernel<<<N_GRID, 256, 0, stream>>>(t3, sln_g, sln_b, gridf, grid_out, N_GRID);
}
